// MultiLatentAttention_51041391346386
// MI455X (gfx1250) — compile-verified, matrix operand order checked by reading, not run on hardware
//
#include <hip/hip_runtime.h>

// ---------------------------------------------------------------------------
// CDNA5 (gfx1250) MLA implementation: bf16 WMMA GEMMs + flash attention,
// with async global->LDS tile staging (ASYNCcnt path).
// ---------------------------------------------------------------------------

typedef __attribute__((ext_vector_type(16))) __bf16 v16bf;
typedef __attribute__((ext_vector_type(8)))  float  v8f;

struct __align__(16) bf16x8 { __bf16 e[8]; };

__device__ __forceinline__ v8f zero_v8f() {
  v8f z;
#pragma unroll
  for (int i = 0; i < 8; ++i) z[i] = 0.0f;
  return z;
}

// Build a 16x32 (A) or 32x16 (B) bf16 fragment from two contiguous 8-element
// chunks per lane (ISA 7.12.2: lanes 0-15 hold K 0..7 / 16..23 in VGPR 0-3 /
// 4-7; lanes 16-31 hold K 8..15 / 24..31).
__device__ __forceinline__ v16bf make_frag(const __bf16* p0, const __bf16* p1) {
  bf16x8 a = *reinterpret_cast<const bf16x8*>(p0);
  bf16x8 b = *reinterpret_cast<const bf16x8*>(p1);
  v16bf r;
#pragma unroll
  for (int i = 0; i < 8; ++i) { r[i] = a.e[i]; r[i + 8] = b.e[i]; }
  return r;
}

__device__ __forceinline__ v8f wmma_bf16(v16bf a, v16bf b, v8f c) {
  return __builtin_amdgcn_wmma_f32_16x16x32_bf16(
      /*neg_a=*/false, a, /*neg_b=*/false, b,
      /*c_mod=*/(short)0, c, /*reuse_a=*/false, /*reuse_b=*/false);
}

// Async global -> LDS copy of 16 bytes per lane (GLOBAL_LOAD_ASYNC_TO_LDS_B128,
// GV addressing). LDS destination offset = low 32 bits of the flat shared
// address (ISA 10.2: LDS aperture keeps the LDS byte offset in addr[31:0]).
__device__ __forceinline__ void async_ld_b128(void* lds_dst, const void* src) {
  unsigned lds_off = (unsigned)(unsigned long long)lds_dst;
  asm volatile("global_load_async_to_lds_b128 %0, %1, off"
               :
               : "v"(lds_off), "v"(src)
               : "memory");
}

__device__ __forceinline__ void wait_async0() {
  asm volatile("s_wait_asynccnt 0x0" ::: "memory");
}

// ---------------------------------------------------------------------------
// f32 -> bf16 elementwise convert
// ---------------------------------------------------------------------------
__global__ void cvt_f32_bf16(const float* __restrict__ in,
                             __bf16* __restrict__ out, long n) {
  long i = (long)blockIdx.x * blockDim.x + threadIdx.x;
  long stride = (long)gridDim.x * blockDim.x;
  for (; i < n; i += stride) out[i] = (__bf16)in[i];
}

// ---------------------------------------------------------------------------
// GEMM: C[M,N] = A[M,K] * B[N,K]^T   (A,B bf16, C f32 or bf16)
// Block = 256 threads (8 waves). Workgroup tile 128x128, wave tile 32x64.
// K step 32. M must be multiple of 128, K multiple of 32. N guarded.
// Tiles staged with async global->LDS copies (ASYNCcnt).
// ---------------------------------------------------------------------------
__global__ __launch_bounds__(256) void gemm_bf16_wmma(
    const __bf16* __restrict__ A, const __bf16* __restrict__ B,
    void* __restrict__ Cout, int M, int N, int K, int write_bf16) {
  __shared__ __bf16 As[128][40];
  __shared__ __bf16 Bs[128][40];

  const int tid = threadIdx.x;
  const int m0 = blockIdx.y * 128;
  const int n0 = blockIdx.x * 128;
  const int w = tid >> 5, lane = tid & 31;
  const int wm = (w >> 1) * 32;   // wave m offset in tile
  const int wn = (w & 1) * 64;    // wave n offset in tile
  const int row = lane & 15, hi = lane >> 4;
  const int sel0 = hi ? 8 : 0;

  v8f acc[2][4];
#pragma unroll
  for (int mb = 0; mb < 2; ++mb)
#pragma unroll
    for (int nb = 0; nb < 4; ++nb) acc[mb][nb] = zero_v8f();

  for (int k0 = 0; k0 < K; k0 += 32) {
    // async cooperative tile staging: 2 x 16B chunks per thread for A and B
    for (int ch = tid; ch < 512; ch += 256) {
      int r = ch >> 2;
      int c8 = (ch & 3) << 3;
      const __bf16* srcA = A + (size_t)(m0 + r) * K + k0 + c8;
      async_ld_b128(&As[r][c8], srcA);
      if (k0 + 32 < K) __builtin_prefetch(srcA + 32, 0, 1);
      // Clamp the N edge: columns >= N produce garbage accumulators that are
      // never stored (per-column independence of D = A*B).
      int rn = n0 + r;
      if (rn >= N) rn = N - 1;
      const __bf16* srcB = B + (size_t)rn * K + k0 + c8;
      async_ld_b128(&Bs[r][c8], srcB);
      if (k0 + 32 < K) __builtin_prefetch(srcB + 32, 0, 1);
    }
    wait_async0();
    __syncthreads();

    v16bf af[2], bfr[4];
#pragma unroll
    for (int mb = 0; mb < 2; ++mb) {
      const __bf16* p = &As[wm + mb * 16 + row][0];
      af[mb] = make_frag(p + sel0, p + sel0 + 16);
    }
#pragma unroll
    for (int nb = 0; nb < 4; ++nb) {
      const __bf16* p = &Bs[wn + nb * 16 + row][0];
      bfr[nb] = make_frag(p + sel0, p + sel0 + 16);
    }
#pragma unroll
    for (int mb = 0; mb < 2; ++mb)
#pragma unroll
      for (int nb = 0; nb < 4; ++nb)
        acc[mb][nb] = wmma_bf16(af[mb], bfr[nb], acc[mb][nb]);
    __syncthreads();
  }

  // Epilogue: C/D layout = lane (0-15|16-31) -> col, VGPR j -> row j + 8*hi.
#pragma unroll
  for (int mb = 0; mb < 2; ++mb)
#pragma unroll
    for (int nb = 0; nb < 4; ++nb)
#pragma unroll
      for (int j = 0; j < 8; ++j) {
        int r = m0 + wm + mb * 16 + j + 8 * hi;
        int c = n0 + wn + nb * 16 + row;
        if (c < N) {
          float v = acc[mb][nb][j];
          if (write_bf16)
            ((__bf16*)Cout)[(size_t)r * N + c] = (__bf16)v;
          else
            ((float*)Cout)[(size_t)r * N + c] = v;
        }
      }
}

// ---------------------------------------------------------------------------
// RoPE: in (rows, heads*64) f32 -> out (rows, heads*64) bf16
// half-split rotation as in the reference (d=64, half=32)
// ---------------------------------------------------------------------------
__global__ void rope_f32_bf16(const float* __restrict__ in,
                              __bf16* __restrict__ out,
                              const int* __restrict__ positions,
                              int rows, int heads, int S) {
  int total = rows * heads * 32;
  for (int idx = blockIdx.x * blockDim.x + threadIdx.x; idx < total;
       idx += gridDim.x * blockDim.x) {
    int i = idx & 31;
    int hh = (idx >> 5) % heads;
    int rw = idx / (32 * heads);
    int pos = positions[rw % S];
    float inv = __powf(10000.0f, -(float)i * (1.0f / 32.0f));
    float ang = (float)pos * inv;
    float sn, cs;
    __sincosf(ang, &sn, &cs);
    size_t base = (size_t)rw * heads * 64 + (size_t)hh * 64;
    float t1 = in[base + i];
    float t2 = in[base + 32 + i];
    out[base + i]      = (__bf16)(t1 * cs - t2 * sn);
    out[base + 32 + i] = (__bf16)(t2 * cs + t1 * sn);
  }
}

// ---------------------------------------------------------------------------
// Flash attention (causal). One wave handles 16 query rows of one (b,h).
// qc: (B*S, H*128) bf16   q nope
// qr: (B*S, H*64)  bf16   q rope (already rotated)
// kc: (B*S, H*128) bf16   k nope
// kr: (B*S, 64)    bf16   k rope (already rotated, shared across heads)
// vc: (B*S, H*128) bf16
// ao: (B*S, H*128) bf16   output
// ---------------------------------------------------------------------------
#define MLA_H 16
__global__ __launch_bounds__(128) void mla_attn(
    const __bf16* __restrict__ qc, const __bf16* __restrict__ qr,
    const __bf16* __restrict__ kc, const __bf16* __restrict__ kr,
    const __bf16* __restrict__ vc, __bf16* __restrict__ ao,
    int S, float scale) {
  __shared__ __bf16 P[4][16 * 32];  // per-wave P-transpose staging

  const int w = threadIdx.x >> 5, lane = threadIdx.x & 31;
  const int nqb = S >> 4;
  int gw = blockIdx.x * 4 + w;
  const int qb = gw % nqb; gw /= nqb;
  const int h = gw % MLA_H; gw /= MLA_H;
  const int b = gw;
  const int q0 = qb << 4;
  const int row = lane & 15, hi = lane >> 4, sel0 = hi ? 8 : 0;

  // Resident Q fragments: 6 chunks of K=32 over d = 128 (nope) + 64 (rope).
  const size_t xrow = (size_t)(b * S + q0 + row);
  v16bf qf[6];
#pragma unroll
  for (int c = 0; c < 4; ++c) {
    const __bf16* p = qc + xrow * (MLA_H * 128) + h * 128 + c * 32;
    qf[c] = make_frag(p + sel0, p + sel0 + 16);
  }
#pragma unroll
  for (int c = 0; c < 2; ++c) {
    const __bf16* p = qr + xrow * (MLA_H * 64) + h * 64 + c * 32;
    qf[4 + c] = make_frag(p + sel0, p + sel0 + 16);
  }

  v8f o[8];
#pragma unroll
  for (int nb = 0; nb < 8; ++nb) o[nb] = zero_v8f();
  float m[8], l[8];
#pragma unroll
  for (int j = 0; j < 8; ++j) { m[j] = -1e30f; l[j] = 0.0f; }

  __bf16* Pw = &P[w][0];
  const int kmaxrow = q0 + 15;

  for (int kt = 0; 32 * kt <= kmaxrow; ++kt) {
    // ---- scores: two 16x16 blocks over a 32-key tile ----
    v8f s0 = zero_v8f(), s1 = zero_v8f();
    {
      int kn = kt * 32 + row;
      int knc = kn < S ? kn : S - 1;
      size_t krow = (size_t)(b * S + knc);
#pragma unroll
      for (int c = 0; c < 4; ++c) {
        const __bf16* p = kc + krow * (MLA_H * 128) + h * 128 + c * 32;
        s0 = wmma_bf16(qf[c], make_frag(p + sel0, p + sel0 + 16), s0);
      }
#pragma unroll
      for (int c = 0; c < 2; ++c) {
        const __bf16* p = kr + krow * 64 + c * 32;
        s0 = wmma_bf16(qf[4 + c], make_frag(p + sel0, p + sel0 + 16), s0);
      }
    }
    {
      int kn = kt * 32 + 16 + row;
      int knc = kn < S ? kn : S - 1;
      size_t krow = (size_t)(b * S + knc);
#pragma unroll
      for (int c = 0; c < 4; ++c) {
        const __bf16* p = kc + krow * (MLA_H * 128) + h * 128 + c * 32;
        s1 = wmma_bf16(qf[c], make_frag(p + sel0, p + sel0 + 16), s1);
      }
#pragma unroll
      for (int c = 0; c < 2; ++c) {
        const __bf16* p = kr + krow * 64 + c * 32;
        s1 = wmma_bf16(qf[4 + c], make_frag(p + sel0, p + sel0 + 16), s1);
      }
    }

    // ---- mask + online softmax (row = j + 8*hi, col = lane&15) ----
    const int kc0 = kt * 32 + row;
    const int kc1 = kc0 + 16;
#pragma unroll
    for (int j = 0; j < 8; ++j) {
      int qrw = q0 + j + 8 * hi;
      float v0 = (kc0 <= qrw) ? s0[j] * scale : -1e30f;
      float v1 = (kc1 <= qrw) ? s1[j] * scale : -1e30f;
      float mx = fmaxf(v0, v1);
#pragma unroll
      for (int d = 1; d < 16; d <<= 1) mx = fmaxf(mx, __shfl_xor(mx, d, 32));
      float mnew = fmaxf(m[j], mx);
      float alpha = __expf(m[j] - mnew);
      m[j] = mnew;
      float p0 = __expf(v0 - mnew);
      float p1 = __expf(v1 - mnew);
      float rs = p0 + p1;
#pragma unroll
      for (int d = 1; d < 16; d <<= 1) rs += __shfl_xor(rs, d, 32);
      l[j] = l[j] * alpha + rs;
#pragma unroll
      for (int nb = 0; nb < 8; ++nb) o[nb][j] *= alpha;
      // stage P (16 rows x 32 keys) for the A-fragment transpose
      Pw[(j + 8 * hi) * 32 + row] = (__bf16)p0;
      Pw[(j + 8 * hi) * 32 + 16 + row] = (__bf16)p1;
    }
    asm volatile("s_wait_dscnt 0x0" ::: "memory");

    // ---- P @ V : one A fragment (16x32), 8 B fragments (32x16 each) ----
    v16bf pf = make_frag(&Pw[row * 32 + sel0], &Pw[row * 32 + sel0 + 16]);
#pragma unroll
    for (int nb = 0; nb < 8; ++nb) {
      v16bf vf;
      int vcol = h * 128 + nb * 16 + row;
#pragma unroll
      for (int i = 0; i < 8; ++i) {
        int k0i = kt * 32 + sel0 + i;       if (k0i >= S) k0i = S - 1;
        int k1i = kt * 32 + sel0 + 16 + i;  if (k1i >= S) k1i = S - 1;
        vf[i]     = vc[(size_t)(b * S + k0i) * (MLA_H * 128) + vcol];
        vf[i + 8] = vc[(size_t)(b * S + k1i) * (MLA_H * 128) + vcol];
      }
      o[nb] = wmma_bf16(pf, vf, o[nb]);
    }
  }

  // ---- epilogue: divide by l, write bf16 ----
#pragma unroll
  for (int nb = 0; nb < 8; ++nb)
#pragma unroll
    for (int j = 0; j < 8; ++j) {
      int r = q0 + j + 8 * hi;
      int cidx = h * 128 + nb * 16 + row;
      float v = o[nb][j] / l[j];
      ao[(size_t)(b * S + r) * (MLA_H * 128) + cidx] = (__bf16)v;
    }
}

// ---------------------------------------------------------------------------
// Host launcher
// ---------------------------------------------------------------------------
extern "C" void kernel_launch(void* const* d_in, const int* in_sizes, int n_in,
                              void* d_out, int out_size, void* d_ws,
                              size_t ws_size, hipStream_t stream) {
  (void)in_sizes; (void)n_in; (void)out_size; (void)ws_size;
  const int Bb = 2, S = 2048, D = 2048, H = 16;
  const int QR = 1536, KVR = 512, NOPE = 128, RD = 64, VD = 128;
  const int M = Bb * S;  // 4096
  const float scale = 0.0721687836487032f;  // 1/sqrt(192)

  const float* x        = (const float*)d_in[0];
  const int*   posns    = (const int*)d_in[1];
  const float* Wq_down  = (const float*)d_in[2];
  const float* Wq_up    = (const float*)d_in[3];
  const float* Wq_rope  = (const float*)d_in[4];
  const float* Wkv_down = (const float*)d_in[5];
  const float* Wk_up    = (const float*)d_in[6];
  const float* Wv_up    = (const float*)d_in[7];
  const float* Wk_rope  = (const float*)d_in[8];
  const float* Wo       = (const float*)d_in[9];
  float* out = (float*)d_out;

  char* ws = (char*)d_ws;
  size_t off = 0;
  auto alloc = [&](size_t bytes) -> void* {
    void* p = ws + off;
    off = (off + bytes + 255) & ~(size_t)255;
    return p;
  };

  __bf16* xb   = (__bf16*)alloc((size_t)M * D * 2);
  __bf16* wqd  = (__bf16*)alloc((size_t)QR * D * 2);
  __bf16* wqu  = (__bf16*)alloc((size_t)H * NOPE * QR * 2);
  __bf16* wqr  = (__bf16*)alloc((size_t)H * RD * QR * 2);
  __bf16* wkvd = (__bf16*)alloc((size_t)KVR * D * 2);
  __bf16* wku  = (__bf16*)alloc((size_t)H * NOPE * KVR * 2);
  __bf16* wvu  = (__bf16*)alloc((size_t)H * VD * KVR * 2);
  __bf16* wkr  = (__bf16*)alloc((size_t)RD * D * 2);
  __bf16* wo   = (__bf16*)alloc((size_t)D * H * VD * 2);
  __bf16* cqb  = (__bf16*)alloc((size_t)M * QR * 2);
  __bf16* ckvb = (__bf16*)alloc((size_t)M * KVR * 2);
  __bf16* qcb  = (__bf16*)alloc((size_t)M * H * NOPE * 2);
  float*  qrf  = (float*)alloc((size_t)M * H * RD * 4);
  __bf16* kcb  = (__bf16*)alloc((size_t)M * H * NOPE * 2);
  __bf16* vcb  = (__bf16*)alloc((size_t)M * H * VD * 2);
  float*  krf  = (float*)alloc((size_t)M * RD * 4);
  __bf16* qr2  = (__bf16*)alloc((size_t)M * H * RD * 2);
  __bf16* kr2  = (__bf16*)alloc((size_t)M * RD * 2);
  __bf16* aob  = (__bf16*)alloc((size_t)M * H * VD * 2);

  auto cvt = [&](const float* src, __bf16* dst, long n) {
    int blocks = (int)((n + 255) / 256);
    if (blocks > 2048) blocks = 2048;
    cvt_f32_bf16<<<blocks, 256, 0, stream>>>(src, dst, n);
  };
  cvt(x, xb, (long)M * D);
  cvt(Wq_down, wqd, (long)QR * D);
  cvt(Wq_up, wqu, (long)H * NOPE * QR);
  cvt(Wq_rope, wqr, (long)H * RD * QR);
  cvt(Wkv_down, wkvd, (long)KVR * D);
  cvt(Wk_up, wku, (long)H * NOPE * KVR);
  cvt(Wv_up, wvu, (long)H * VD * KVR);
  cvt(Wk_rope, wkr, (long)RD * D);
  cvt(Wo, wo, (long)D * H * VD);

  auto gemm = [&](const __bf16* A, const __bf16* B, void* C, int m, int n,
                  int k, int wbf16) {
    dim3 grid((n + 127) / 128, (m + 127) / 128);
    gemm_bf16_wmma<<<grid, 256, 0, stream>>>(A, B, C, m, n, k, wbf16);
  };

  gemm(xb, wqd, cqb, M, QR, D, 1);                 // c_q
  gemm(cqb, wqu, qcb, M, H * NOPE, QR, 1);         // q_c
  gemm(cqb, wqr, qrf, M, H * RD, QR, 0);           // q_r (f32 for rope)
  gemm(xb, wkvd, ckvb, M, KVR, D, 1);              // c_kv
  gemm(ckvb, wku, kcb, M, H * NOPE, KVR, 1);       // k_c
  gemm(ckvb, wvu, vcb, M, H * VD, KVR, 1);         // v_c
  gemm(xb, wkr, krf, M, RD, D, 0);                 // k_r (f32 for rope)

  rope_f32_bf16<<<2048, 256, 0, stream>>>(qrf, qr2, posns, M, H, S);
  rope_f32_bf16<<<512, 256, 0, stream>>>(krf, kr2, posns, M, 1, S);

  // one wave per 16 q-rows per (b,h): B*H*(S/16) waves, 4 waves per block
  int attn_blocks = Bb * H * (S / 16) / 4;  // 1024
  mla_attn<<<attn_blocks, 128, 0, stream>>>(qcb, qr2, kcb, kr2, vcb, aob, S,
                                            scale);

  gemm(aob, wo, out, M, D, H * VD, 0);             // final @ Wo.T
}
